// LGNN_56461640073736
// MI455X (gfx1250) — compile-verified, matrix operand order checked by reading, not run on hardware
//
#include <hip/hip_runtime.h>
#include <hip/hip_bf16.h>
#include <stdint.h>

#define N_NODES_C 10000
#define N_EDGES_C 80000
#define E_G_C     80000
#define E_LG_C    160000
#define H_C       512
#define K_C       4
#define DH_C      128
#define FF_C      2048
#define GEMM_NT   8

typedef __attribute__((ext_vector_type(16))) __bf16 bf16x16;
typedef __attribute__((ext_vector_type(8)))  float  f32x8;

union FragU { bf16x16 v; uint4 u[2]; };

__device__ __forceinline__ unsigned short f2bf(float f) {
  unsigned u = __float_as_uint(f);
  unsigned r = u + 0x7FFFu + ((u >> 16) & 1u);   // round-to-nearest-even
  return (unsigned short)(r >> 16);
}
__device__ __forceinline__ unsigned encf(float f) {   // order-preserving f32 -> u32
  unsigned u = __float_as_uint(f);
  return (u & 0x80000000u) ? ~u : (u | 0x80000000u);
}
__device__ __forceinline__ float decf(unsigned u) {
  unsigned v = (u & 0x80000000u) ? (u & 0x7FFFFFFFu) : ~u;
  return __uint_as_float(v);
}
// per-lane K index inside a 16x32 bf16 A/B fragment (ISA 7.12.2 layout)
__device__ __forceinline__ int kmap(int e, int half) {
  return (e < 8 ? e : e + 8) + 8 * half;
}

// ---------------- utility kernels ----------------
__global__ void fill_f32_k(float* p, float v, size_t n) {
  size_t i = (size_t)blockIdx.x * blockDim.x + threadIdx.x;
  if (i < n) p[i] = v;
}
__global__ void fill_u32_k(unsigned* p, unsigned v, size_t n) {
  size_t i = (size_t)blockIdx.x * blockDim.x + threadIdx.x;
  if (i < n) p[i] = v;
}
__global__ void cvt_bf_k(const float* __restrict__ in, unsigned short* __restrict__ out, size_t n) {
  size_t i = (size_t)blockIdx.x * blockDim.x + threadIdx.x;
  if (i < n) out[i] = f2bf(in[i]);
}
// pack [n,512] f32 into bf16 columns [dstOff, dstOff+512) of an [n,1024] buffer
__global__ void cvt_bf_strided_k(const float* __restrict__ in, unsigned short* __restrict__ out,
                                 int n, int dstOff) {
  size_t i = (size_t)blockIdx.x * blockDim.x + threadIdx.x;
  if (i >= (size_t)n * H_C) return;
  size_t row = i >> 9; int col = (int)(i & (H_C - 1));
  out[row * (size_t)(2 * H_C) + dstOff + col] = f2bf(in[i]);
}
// lg_x = rel_embed[edge_feat]
__global__ void relgather_k(const float* __restrict__ rel, const int* __restrict__ ef,
                            float* __restrict__ lg) {
  size_t i = (size_t)blockIdx.x * blockDim.x + threadIdx.x;
  if (i >= (size_t)N_EDGES_C * H_C) return;
  lg[i] = rel[(size_t)ef[i >> 9] * H_C + (i & (H_C - 1))];
}
// gat_out init = bias broadcast (gat_b flat [K*DH] == column index & 511)
__global__ void init_gatout_k(const float* __restrict__ gb, float* __restrict__ gout, size_t n) {
  size_t i = (size_t)blockIdx.x * blockDim.x + threadIdx.x;
  if (i < n) gout[i] = gb[i & (H_C - 1)];
}

// ---------------- weight packing (f32 -> fragment-native bf16) ----------------
// Bp[((nTile*KT + kT)*32 + lane)*16 + e] = bf16(W[kT*32 + kmap(e,half)][nTile*16 + lane&15])
// gat==1: W is gat_W[l,c] of shape [4 heads,512,128] concatenated along columns
__global__ void pack_B_k(const float* __restrict__ W, unsigned short* __restrict__ Bp,
                         int Kd, int Nd, int gat) {
  size_t i = (size_t)blockIdx.x * blockDim.x + threadIdx.x;
  if (i >= (size_t)Kd * Nd) return;
  int e    = (int)(i & 15);
  int lane = (int)((i >> 4) & 31);
  size_t tile = i >> 9;
  int KT = Kd >> 5;
  int nTile = (int)(tile / KT), kT = (int)(tile % KT);
  int half = lane >> 4;
  int k = kT * 32 + kmap(e, half);
  int n = nTile * 16 + (lane & 15);
  float v;
  if (gat) v = W[((size_t)(n >> 7) * H_C + k) * DH_C + (n & (DH_C - 1))];
  else     v = W[(size_t)k * Nd + n];
  Bp[i] = f2bf(v);
}

// ---------------- WMMA bf16 GEMM: C[M,Nd] = A[M,Kd] @ B + bias (opt relu, opt bf16 copy) ---
// block = 256 (8 waves). All 8 waves share one 8-tile (32x128) B panel per K-step:
// stage it in LDS (double-buffered, 2x8KB) so the panel is fetched from global ONCE per
// block, next panel's global loads overlap current WMMAs fed from ds_load_b128.
// Each wave owns a 16x128 output strip (8 f32x8 accumulators).
__global__ __launch_bounds__(256)
void gemm_bf16_k(const unsigned short* __restrict__ A, const unsigned short* __restrict__ Bp,
                 const float* __restrict__ bias, float* __restrict__ C,
                 unsigned short* __restrict__ Cbf, int M, int Kd, int Nd, int relu) {
  __shared__ uint4 ldsB[2][GEMM_NT * 64];          // 2 buffers x 8 tiles x 1KB
  int tid  = threadIdx.x;
  int lane = tid & 31;
  int wave = tid >> 5;
  int half = lane >> 4;
  int l16  = lane & 15;
  int rowTile = blockIdx.y * 8 + wave;
  bool active = (rowTile * 16) < M;                // wave-uniform; all waves hit barriers
  int row = active ? rowTile * 16 + l16 : l16;
  int nBase = blockIdx.x * GEMM_NT;
  int KT = Kd >> 5;

  // cooperative staging role: thread tid copies 32B of tile (tid>>5), sub-lane (tid&31)
  int sTile = tid >> 5;
  int sLane = tid & 31;
  const uint4* bSrc = (const uint4*)(Bp + (size_t)(nBase + sTile) * KT * 512) + sLane * 2;
  int sIdx = sTile * 64 + sLane * 2;

  f32x8 acc[GEMM_NT];
#pragma unroll
  for (int t = 0; t < GEMM_NT; ++t) acc[t] = (f32x8){0.f, 0.f, 0.f, 0.f, 0.f, 0.f, 0.f, 0.f};

  // stage panel kT=0 into buffer 0  (tile stride in Bp = 512 bf16 = 64 uint4)
  ldsB[0][sIdx]     = bSrc[0];
  ldsB[0][sIdx + 1] = bSrc[1];
  __syncthreads();

  const unsigned short* arow = A + (size_t)row * Kd;
  for (int kT = 0; kT < KT; ++kT) {
    int cur = kT & 1;
    // issue next panel's global loads early (lands in buf 1-cur)
    if (kT + 1 < KT) {
      const uint4* s = bSrc + (size_t)(kT + 1) * 32;   // +32 uint4 = 256 bf16 per kT
      ldsB[1 - cur][sIdx]     = s[0];
      ldsB[1 - cur][sIdx + 1] = s[1];
    }
    // A fragment (two contiguous 16B loads per lane; ISA K-mapping)
    FragU a;
    int kk = kT * 32;
    a.u[0] = *(const uint4*)(arow + kk + 8 * half);
    a.u[1] = *(const uint4*)(arow + kk + 16 + 8 * half);
    if (kk + 32 < Kd) __builtin_prefetch(arow + kk + 32 + 8 * half, 0, 0);
    // B fragments from LDS, batched ahead of the WMMA chain
    FragU b[GEMM_NT];
#pragma unroll
    for (int t = 0; t < GEMM_NT; ++t) {
      b[t].u[0] = ldsB[cur][t * 64 + lane * 2];
      b[t].u[1] = ldsB[cur][t * 64 + lane * 2 + 1];
    }
#pragma unroll
    for (int t = 0; t < GEMM_NT; ++t)
      acc[t] = __builtin_amdgcn_wmma_f32_16x16x32_bf16(
          false, a.v, false, b[t].v, (short)0, acc[t], false, false);
    __syncthreads();
  }

  if (active) {
#pragma unroll
    for (int t = 0; t < GEMM_NT; ++t) {
      int colBase = (nBase + t) * 16 + l16;
      float bv = bias ? bias[colBase] : 0.f;
#pragma unroll
      for (int r = 0; r < 8; ++r) {
        int m = rowTile * 16 + r + 8 * half;      // D layout: VGPR r -> M=r (+8 hi half)
        float v = acc[t][r] + bv;
        if (relu) v = v > 0.f ? v : 0.f;
        size_t idx = (size_t)m * Nd + colBase;
        C[idx] = v;
        if (Cbf) Cbf[idx] = f2bf(v);
      }
    }
  }
}

// ---------------- GAT attention pieces ----------------
__global__ void eler_k(const float* __restrict__ h, const float* __restrict__ al,
                       const float* __restrict__ ar, float* __restrict__ el,
                       float* __restrict__ er, int n) {
  int i = blockIdx.x * blockDim.x + threadIdx.x;    // over n*K
  if (i >= n * K_C) return;
  int node = i >> 2, k = i & 3;
  const float* hp = h + (size_t)node * H_C + k * DH_C;
  float sl = 0.f, sr = 0.f;
  for (int d = 0; d < DH_C; ++d) { float hv = hp[d]; sl += hv * al[k * DH_C + d]; sr += hv * ar[k * DH_C + d]; }
  el[i] = sl; er[i] = sr;
}
__global__ void gat_edge1_k(const float* __restrict__ el, const float* __restrict__ er,
                            const int* __restrict__ src, const int* __restrict__ dst,
                            float* __restrict__ eBuf, unsigned* __restrict__ mxEnc, int E) {
  int i = blockIdx.x * blockDim.x + threadIdx.x;    // i = k*E + j
  if (i >= K_C * E) return;
  int k = i / E;
  int s = src[i], d = dst[i];
  float ev = el[s * K_C + k] + er[d * K_C + k];
  ev = ev > 0.f ? ev : 0.2f * ev;                   // LeakyReLU(0.2)
  eBuf[i] = ev;
  atomicMax(&mxEnc[d * K_C + k], encf(ev));
}
__global__ void gat_edge2_k(const float* __restrict__ eBuf, const int* __restrict__ dst,
                            const unsigned* __restrict__ mxEnc, float* __restrict__ wBuf,
                            float* __restrict__ z, int E) {
  int i = blockIdx.x * blockDim.x + threadIdx.x;
  if (i >= K_C * E) return;
  int k = i / E;
  int d = dst[i];
  float w = __expf(eBuf[i] - decf(mxEnc[d * K_C + k]));
  wBuf[i] = w;
  atomicAdd(&z[d * K_C + k], w);
}
__global__ __launch_bounds__(256)
void gat_scatter_k(const float* __restrict__ h, const float* __restrict__ wBuf,
                   const float* __restrict__ z, const int* __restrict__ src,
                   const int* __restrict__ dst, float* __restrict__ gout, int E) {
  int widx = blockIdx.x * (blockDim.x >> 5) + (threadIdx.x >> 5);   // wave per (k,edge)
  int lane = threadIdx.x & 31;
  if (widx >= K_C * E) return;
  int k = widx / E;
  int s = src[widx], d = dst[widx];
  float alpha = wBuf[widx] / z[d * K_C + k];
  const float* hp = h + (size_t)s * H_C + k * DH_C;
  float* gp = gout + (size_t)d * H_C + k * DH_C;
#pragma unroll
  for (int c = lane; c < DH_C; c += 32) atomicAdd(gp + c, hp[c] * alpha);
}

// ---------------- layer-level aggregations ----------------
__global__ __launch_bounds__(256)
void edge_agg_k(const float* __restrict__ lg, const int* __restrict__ se,
                const int* __restrict__ de, float* __restrict__ agg) {
  int j = blockIdx.x * (blockDim.x >> 5) + (threadIdx.x >> 5);      // wave per edge
  int lane = threadIdx.x & 31;
  if (j >= N_EDGES_C) return;
  const float* p = lg + (size_t)j * H_C;
  int s = se[j], d = de[j];
  for (int c = lane; c < H_C; c += 32) {
    float v = p[c];
    atomicAdd(agg + (size_t)s * H_C + c, v);
    atomicAdd(agg + (size_t)d * H_C + c, v);
  }
}
__global__ void node_agg_k(const float* __restrict__ x, const int* __restrict__ se,
                           const int* __restrict__ de, float* __restrict__ nagg) {
  size_t i = (size_t)blockIdx.x * blockDim.x + threadIdx.x;         // over E * H/4
  if (i >= (size_t)N_EDGES_C * (H_C / 4)) return;
  size_t j = i / (H_C / 4);
  int c4 = (int)(i % (H_C / 4));
  const float4 a = ((const float4*)(x + (size_t)se[j] * H_C))[c4];
  const float4 b = ((const float4*)(x + (size_t)de[j] * H_C))[c4];
  float4 o; o.x = a.x + b.x; o.y = a.y + b.y; o.z = a.z + b.z; o.w = a.w + b.w;
  ((float4*)(nagg + j * H_C))[c4] = o;
}

// ---------------- LayerNorm (wave per row, wave32 shuffle reduce) ----------------
__global__ __launch_bounds__(256)
void ln_k(const float* __restrict__ a, const float* __restrict__ bnd,
          const float* __restrict__ g, const float* __restrict__ be,
          float* __restrict__ out, unsigned short* __restrict__ outBf, int n) {
  int row  = blockIdx.x * 8 + (threadIdx.x >> 5);
  int lane = threadIdx.x & 31;
  if (row >= n) return;
  const float* pa = a + (size_t)row * H_C;
  const float* pb = bnd + (size_t)row * H_C;
  float v[16];
  float s = 0.f, s2 = 0.f;
#pragma unroll
  for (int t = 0; t < 16; ++t) {
    int c = lane * 16 + t;
    float xv = pa[c] + pb[c];
    v[t] = xv; s += xv; s2 += xv * xv;
  }
  for (int off = 16; off; off >>= 1) {
    s  += __shfl_xor(s,  off, 32);
    s2 += __shfl_xor(s2, off, 32);
  }
  float m  = s * (1.f / H_C);
  float rs = rsqrtf(s2 * (1.f / H_C) - m * m + 1e-5f);
#pragma unroll
  for (int t = 0; t < 16; ++t) {
    int c = lane * 16 + t;
    float o = (v[t] - m) * rs * g[c] + be[c];
    out[(size_t)row * H_C + c] = o;
    if (outBf) outBf[(size_t)row * H_C + c] = f2bf(o);
  }
}

// ---------------- host orchestration ----------------
struct Scratch {
  float *lgA, *lgB, *xB, *edge_agg, *node_agg;
  unsigned short *featBf, *fuseIn, *oBf, *ff1Bf;
  float *h, *gat_out, *fuseOut, *o, *ff1, *ff2;
  float *el, *er, *eBuf, *wBuf, *z;
  unsigned *mxEnc;
  unsigned short *gatWp, *fuseWp, *w1p, *w2p;
};

static inline unsigned gb1(size_t total, int block) { return (unsigned)((total + block - 1) / block); }

static void run_core(hipStream_t stream, const Scratch& S,
                     const float* feat, const float* agg,
                     const int* src, const int* dst, int n, int E,
                     const float* gW, const float* gal, const float* gar, const float* gb,
                     const float* fW, const float* fb,
                     const float* w1, const float* b1,
                     const float* w2, const float* b2,
                     const float* g1, const float* be1,
                     const float* g2, const float* be2,
                     float* out) {
  const int T = 256;
  size_t nH = (size_t)n * H_C;
  // h = feat @ concat_k(W_k)  (bf16 WMMA)
  cvt_bf_k<<<gb1(nH, T), T, 0, stream>>>(feat, S.featBf, nH);
  pack_B_k<<<gb1((size_t)H_C * H_C, T), T, 0, stream>>>(gW, S.gatWp, H_C, H_C, 1);
  {
    dim3 g(H_C / 128, (unsigned)(((n + 15) / 16 + 7) / 8));
    gemm_bf16_k<<<g, 256, 0, stream>>>(S.featBf, S.gatWp, nullptr, S.h, nullptr, n, H_C, H_C, 0);
  }
  // attention logits + segment softmax + scatter
  eler_k<<<gb1((size_t)n * K_C, T), T, 0, stream>>>(S.h, gal, gar, S.el, S.er, n);
  init_gatout_k<<<gb1(nH, T), T, 0, stream>>>(gb, S.gat_out, nH);
  fill_u32_k<<<gb1((size_t)n * K_C, T), T, 0, stream>>>(S.mxEnc, 0u, (size_t)n * K_C);
  fill_f32_k<<<gb1((size_t)n * K_C, T), T, 0, stream>>>(S.z, 0.f, (size_t)n * K_C);
  size_t KE = (size_t)K_C * E;
  gat_edge1_k<<<gb1(KE, T), T, 0, stream>>>(S.el, S.er, src, dst, S.eBuf, S.mxEnc, E);
  gat_edge2_k<<<gb1(KE, T), T, 0, stream>>>(S.eBuf, dst, S.mxEnc, S.wBuf, S.z, E);
  gat_scatter_k<<<gb1(KE, 8), 256, 0, stream>>>(S.h, S.wBuf, S.z, src, dst, S.gat_out, E);
  // fuse = [kh, agg] @ fuse_W + fb
  cvt_bf_strided_k<<<gb1(nH, T), T, 0, stream>>>(S.gat_out, S.fuseIn, n, 0);
  cvt_bf_strided_k<<<gb1(nH, T), T, 0, stream>>>(agg, S.fuseIn, n, H_C);
  pack_B_k<<<gb1((size_t)2 * H_C * H_C, T), T, 0, stream>>>(fW, S.fuseWp, 2 * H_C, H_C, 0);
  {
    dim3 g(H_C / 128, (unsigned)(((n + 15) / 16 + 7) / 8));
    gemm_bf16_k<<<g, 256, 0, stream>>>(S.fuseIn, S.fuseWp, fb, S.fuseOut, nullptr, n, 2 * H_C, H_C, 0);
  }
  // o = LN(feat + fuse)
  ln_k<<<gb1((size_t)n, 8), 256, 0, stream>>>(feat, S.fuseOut, g1, be1, S.o, S.oBf, n);
  // ff = relu(o @ W1 + b1) @ W2 + b2
  pack_B_k<<<gb1((size_t)H_C * FF_C, T), T, 0, stream>>>(w1, S.w1p, H_C, FF_C, 0);
  {
    dim3 g(FF_C / 128, (unsigned)(((n + 15) / 16 + 7) / 8));
    gemm_bf16_k<<<g, 256, 0, stream>>>(S.oBf, S.w1p, b1, S.ff1, S.ff1Bf, n, H_C, FF_C, 1);
  }
  pack_B_k<<<gb1((size_t)FF_C * H_C, T), T, 0, stream>>>(w2, S.w2p, FF_C, H_C, 0);
  {
    dim3 g(H_C / 128, (unsigned)(((n + 15) / 16 + 7) / 8));
    gemm_bf16_k<<<g, 256, 0, stream>>>(S.ff1Bf, S.w2p, b2, S.ff2, nullptr, n, FF_C, H_C, 0);
  }
  // out = LN(o + ff)
  ln_k<<<gb1((size_t)n, 8), 256, 0, stream>>>(S.o, S.ff2, g2, be2, out, nullptr, n);
}

extern "C" void kernel_launch(void* const* d_in, const int* in_sizes, int n_in,
                              void* d_out, int out_size, void* d_ws, size_t ws_size,
                              hipStream_t stream) {
  (void)in_sizes; (void)n_in; (void)out_size; (void)ws_size;
  const float* x      = (const float*)d_in[0];
  const float* rel    = (const float*)d_in[1];
  const float* gat_W  = (const float*)d_in[2];
  const float* gat_al = (const float*)d_in[3];
  const float* gat_ar = (const float*)d_in[4];
  const float* gat_b  = (const float*)d_in[5];
  const float* fuse_W = (const float*)d_in[6];
  const float* fuse_b = (const float*)d_in[7];
  const float* ff_W1  = (const float*)d_in[8];
  const float* ff_b1  = (const float*)d_in[9];
  const float* ff_W2  = (const float*)d_in[10];
  const float* ff_b2  = (const float*)d_in[11];
  const float* ln1_g  = (const float*)d_in[12];
  const float* ln1_b  = (const float*)d_in[13];
  const float* ln2_g  = (const float*)d_in[14];
  const float* ln2_b  = (const float*)d_in[15];
  const int* edge_feat = (const int*)d_in[16];
  const int* src_g  = (const int*)d_in[17];
  const int* dst_g  = (const int*)d_in[18];
  const int* src_lg = (const int*)d_in[19];
  const int* dst_lg = (const int*)d_in[20];
  const int* src_e  = (const int*)d_in[21];
  const int* dst_e  = (const int*)d_in[22];

  char* base = (char*)d_ws;
  size_t off = 0;
  auto carve = [&](size_t bytes) -> void* {
    void* p = base + off;
    off += (bytes + 255) & ~(size_t)255;
    return p;
  };
  Scratch S;
  const size_t EH = (size_t)N_EDGES_C * H_C;       // max-n row buffers
  S.lgA      = (float*)carve(EH * 4);
  S.lgB      = (float*)carve(EH * 4);
  S.xB       = (float*)carve((size_t)N_NODES_C * H_C * 4);
  S.edge_agg = (float*)carve((size_t)N_NODES_C * H_C * 4);
  S.node_agg = (float*)carve(EH * 4);
  S.featBf   = (unsigned short*)carve(EH * 2);
  S.fuseIn   = (unsigned short*)carve(EH * 2 * 2);
  S.oBf      = (unsigned short*)carve(EH * 2);
  S.ff1Bf    = (unsigned short*)carve((size_t)N_EDGES_C * FF_C * 2);
  S.h        = (float*)carve(EH * 4);
  S.gat_out  = (float*)carve(EH * 4);
  S.fuseOut  = (float*)carve(EH * 4);
  S.o        = (float*)carve(EH * 4);
  S.ff1      = (float*)carve((size_t)N_EDGES_C * FF_C * 4);
  S.ff2      = (float*)carve(EH * 4);
  S.el       = (float*)carve((size_t)N_EDGES_C * K_C * 4);
  S.er       = (float*)carve((size_t)N_EDGES_C * K_C * 4);
  S.eBuf     = (float*)carve((size_t)K_C * E_LG_C * 4);
  S.wBuf     = (float*)carve((size_t)K_C * E_LG_C * 4);
  S.z        = (float*)carve((size_t)N_EDGES_C * K_C * 4);
  S.mxEnc    = (unsigned*)carve((size_t)N_EDGES_C * K_C * 4);
  S.gatWp    = (unsigned short*)carve((size_t)H_C * H_C * 2);
  S.fuseWp   = (unsigned short*)carve((size_t)2 * H_C * H_C * 2);
  S.w1p      = (unsigned short*)carve((size_t)H_C * FF_C * 2);
  S.w2p      = (unsigned short*)carve((size_t)FF_C * H_C * 2);

  const int T = 256;
  // lg_x = rel_embed[edge_feat]
  relgather_k<<<gb1(EH, T), T, 0, stream>>>(rel, edge_feat, S.lgA);

  auto P = [&](const float* b, int l, int c, size_t per) { return b + ((size_t)l * 2 + c) * per; };
  float* out_x  = (float*)d_out;
  float* out_lg = out_x + (size_t)N_NODES_C * H_C;

  const float* curX  = x;
  const float* curLg = S.lgA;
  for (int l = 0; l < 2; ++l) {
    // edge_agg = seg_sum(lg_x, src_e) + seg_sum(lg_x, dst_e);  node_agg = x[src_e] + x[dst_e]
    fill_f32_k<<<gb1((size_t)N_NODES_C * H_C, T), T, 0, stream>>>(S.edge_agg, 0.f, (size_t)N_NODES_C * H_C);
    edge_agg_k<<<gb1((size_t)N_EDGES_C, 8), 256, 0, stream>>>(curLg, src_e, dst_e, S.edge_agg);
    node_agg_k<<<gb1((size_t)N_EDGES_C * (H_C / 4), T), T, 0, stream>>>(curX, src_e, dst_e, S.node_agg);

    float* xOut  = (l == 0) ? S.xB  : out_x;
    float* lgOut = (l == 0) ? S.lgB : out_lg;

    run_core(stream, S, curX, S.edge_agg, src_g, dst_g, N_NODES_C, E_G_C,
             P(gat_W, l, 0, (size_t)K_C * H_C * DH_C), P(gat_al, l, 0, H_C), P(gat_ar, l, 0, H_C),
             P(gat_b, l, 0, H_C), P(fuse_W, l, 0, (size_t)2 * H_C * H_C), P(fuse_b, l, 0, H_C),
             P(ff_W1, l, 0, (size_t)H_C * FF_C), P(ff_b1, l, 0, FF_C),
             P(ff_W2, l, 0, (size_t)FF_C * H_C), P(ff_b2, l, 0, H_C),
             P(ln1_g, l, 0, H_C), P(ln1_b, l, 0, H_C), P(ln2_g, l, 0, H_C), P(ln2_b, l, 0, H_C),
             xOut);
    run_core(stream, S, curLg, S.node_agg, src_lg, dst_lg, N_EDGES_C, E_LG_C,
             P(gat_W, l, 1, (size_t)K_C * H_C * DH_C), P(gat_al, l, 1, H_C), P(gat_ar, l, 1, H_C),
             P(gat_b, l, 1, H_C), P(fuse_W, l, 1, (size_t)2 * H_C * H_C), P(fuse_b, l, 1, H_C),
             P(ff_W1, l, 1, (size_t)H_C * FF_C), P(ff_b1, l, 1, FF_C),
             P(ff_W2, l, 1, (size_t)FF_C * H_C), P(ff_b2, l, 1, H_C),
             P(ln1_g, l, 1, H_C), P(ln1_b, l, 1, H_C), P(ln2_g, l, 1, H_C), P(ln2_b, l, 1, H_C),
             lgOut);
    curX = xOut;
    curLg = lgOut;
  }
}
